// Mamba_12747462934750
// MI455X (gfx1250) — compile-verified, matrix operand order checked by reading, not run on hardware
//
#include <hip/hip_runtime.h>

// ---------------------------------------------------------------------------
// Mamba (2-layer) forward for MI455X / gfx1250.
// GEMMs run on v_wmma_f32_16x16x32_f16 (f16 in, f32 accum).  Everything else
// (conv, softplus, selective scan, gating, rmsnorm) is f32 VALU.
// ---------------------------------------------------------------------------

#define L_   2048
#define D_   768
#define DI_  1536
#define XZ_  3072     // 2*DI
#define NR_  80       // DR + 2*N
#define DR_  48
#define NS_  16
#define NV_  50257

typedef _Float16 v16h __attribute__((ext_vector_type(16)));
typedef _Float16 v8h  __attribute__((ext_vector_type(8)));
typedef float    v8f  __attribute__((ext_vector_type(8)));

// ---------------------------------------------------------------------------
// WMMA fragment load: 16x32 f16 tile, one 16-row slab, row-major, stride K.
// Lane n (n = lane&15) supplies row n; kb = 8*(lane>=16).  The 16 halves of
// the lane cover K = kb..kb+7 and kb+16..kb+23  ->  two global_load_b128.
// ---------------------------------------------------------------------------
__device__ __forceinline__ v16h load_frag16(const _Float16* __restrict__ p) {
  v8h lo = *(const v8h*)(p);
  v8h hi = *(const v8h*)(p + 16);
  v16h f;
#pragma unroll
  for (int i = 0; i < 8; ++i) { f[i] = lo[i]; f[i + 8] = hi[i]; }
  return f;
}

// C[M,N] (f32, ldc=N) = A[M,K] (f16, lda=K)  x  W[N,K]^T (f16, ldb=K)
// Block: 256 threads = 8 waves arranged 2(M) x 4(N); wave tile = 32x32.
// Requires: K % 32 == 0, M % 64 == 0.  N ragged (stores guarded).
__global__ __launch_bounds__(256) void gemm_f16_wmma(
    const _Float16* __restrict__ A, const _Float16* __restrict__ B,
    float* __restrict__ C, int M, int N, int K) {
  const int lane = threadIdx.x & 31;
  const int wave = threadIdx.x >> 5;
  const int m0 = blockIdx.y * 64 + (wave >> 2) * 32;
  const int n0 = blockIdx.x * 128 + (wave & 3) * 32;

  const int lhalf = lane & 15;
  const int koff  = (lane >> 4) << 3;

  int am0 = m0 + lhalf;        // M=2048 multiple of 64: always in range
  int am1 = am0 + 16;
  int bn0 = n0 + lhalf;  if (bn0 > N - 1) bn0 = N - 1;   // clamp ragged N
  int bn1 = bn0 + 16;    if (bn1 > N - 1) bn1 = N - 1;

  const _Float16* pa0 = A + (size_t)am0 * K + koff;
  const _Float16* pa1 = A + (size_t)am1 * K + koff;
  const _Float16* pb0 = B + (size_t)bn0 * K + koff;
  const _Float16* pb1 = B + (size_t)bn1 * K + koff;

  v8f acc00 = {}, acc01 = {}, acc10 = {}, acc11 = {};

  for (int k0 = 0; k0 < K; k0 += 32) {
    v16h a0 = load_frag16(pa0 + k0);
    v16h a1 = load_frag16(pa1 + k0);
    v16h b0 = load_frag16(pb0 + k0);
    v16h b1 = load_frag16(pb1 + k0);
    acc00 = __builtin_amdgcn_wmma_f32_16x16x32_f16(false, a0, false, b0,
                                                   (short)0, acc00, false, false);
    acc01 = __builtin_amdgcn_wmma_f32_16x16x32_f16(false, a0, false, b1,
                                                   (short)0, acc01, false, false);
    acc10 = __builtin_amdgcn_wmma_f32_16x16x32_f16(false, a1, false, b0,
                                                   (short)0, acc10, false, false);
    acc11 = __builtin_amdgcn_wmma_f32_16x16x32_f16(false, a1, false, b1,
                                                   (short)0, acc11, false, false);
  }

  // C/D layout: lanes 0-15 -> M rows 0..7 (VGPR v = row v), lanes 16-31 ->
  // rows 8..15; N = lane&15.
  const int nst  = lhalf;
  const int mrow = (lane >> 4) << 3;
#pragma unroll
  for (int t = 0; t < 4; ++t) {
    const v8f* accs[4] = { &acc00, &acc01, &acc10, &acc11 };
    const v8f& a = *accs[t];
    int mt = m0 + ((t >> 1) << 4) + mrow;
    int nt = n0 + ((t & 1) << 4) + nst;
    if (nt < N) {
      float* p = C + (size_t)mt * N + nt;
#pragma unroll
      for (int v = 0; v < 8; ++v) p[(size_t)v * N] = a[v];
    }
  }
}

// ---------------------------------------------------------------------------
// Elementwise / small kernels
// ---------------------------------------------------------------------------
__global__ void embed_kernel(const int* __restrict__ tok,
                             const float* __restrict__ emb,
                             float* __restrict__ x) {
  int t = blockIdx.x * blockDim.x + threadIdx.x;   // L_*D_
  int l = t / D_, d = t - l * D_;
  x[t] = emb[(size_t)tok[l] * D_ + d];
}

__global__ void cvt_f16_kernel(const float* __restrict__ src,
                               _Float16* __restrict__ dst, int n) {
  int t = blockIdx.x * blockDim.x + threadIdx.x;
  if (t < n) dst[t] = (_Float16)src[t];
}

__global__ __launch_bounds__(256) void rmsnorm_f16_kernel(
    const float* __restrict__ x, const float* __restrict__ w,
    _Float16* __restrict__ out) {
  int l = blockIdx.x;
  const float* xr = x + (size_t)l * D_;
  __shared__ float red[8];
  float s = 0.f;
  for (int i = threadIdx.x; i < D_; i += 256) { float v = xr[i]; s += v * v; }
#pragma unroll
  for (int m = 16; m >= 1; m >>= 1) s += __shfl_xor(s, m);
  if ((threadIdx.x & 31) == 0) red[threadIdx.x >> 5] = s;
  __syncthreads();
  if (threadIdx.x == 0) {
    float t = 0.f;
#pragma unroll
    for (int i = 0; i < 8; ++i) t += red[i];
    red[0] = rsqrtf(t / (float)D_ + 1e-5f);
  }
  __syncthreads();
  float r = red[0];
  for (int i = threadIdx.x; i < D_; i += 256)
    out[(size_t)l * D_ + i] = (_Float16)(xr[i] * r * w[i]);
}

__device__ __forceinline__ float silu_f(float v) {
  return v / (1.f + __expf(-v));
}

// depthwise causal conv (K=4) + bias + SiLU.  Input = first DI_ cols of xz.
__global__ void conv_silu_kernel(const float* __restrict__ xz,
                                 const float* __restrict__ w,
                                 const float* __restrict__ b,
                                 float* __restrict__ xs,
                                 _Float16* __restrict__ xh) {
  int t = blockIdx.x * blockDim.x + threadIdx.x;   // L_*DI_
  int c = t % DI_, l = t / DI_;
  float acc = b[c];
#pragma unroll
  for (int k = 0; k < 4; ++k) {
    int ls = l - 3 + k;
    if (ls >= 0) acc += w[c * 4 + k] * xz[(size_t)ls * XZ_ + c];
  }
  float s = silu_f(acc);
  xs[t] = s;
  xh[t] = (_Float16)s;
}

// delta = softplus(dlt @ dt_w^T + dt_b);  dlt = BCd[:, :48]
__global__ void delta_kernel(const float* __restrict__ BCd,
                             const float* __restrict__ dtw,
                             const float* __restrict__ dtb,
                             float* __restrict__ delta) {
  int t = blockIdx.x * blockDim.x + threadIdx.x;   // L_*DI_
  int d = t % DI_, l = t / DI_;
  const float* bc = BCd + (size_t)l * NR_;
  const float* wr = dtw + (size_t)d * DR_;
  float s = dtb[d];
#pragma unroll 8
  for (int r = 0; r < DR_; ++r) s += bc[r] * wr[r];
  delta[t] = (s > 20.f) ? s : log1pf(__expf(s));
}

// selective scan: lane handles one (channel d, state n); 16-lane xor-shuffle
// reduction (wave32: masks 1/2/4/8 stay inside each 16-lane half).
__global__ __launch_bounds__(256) void scan_kernel(
    const float* __restrict__ delta, const float* __restrict__ xs,
    const float* __restrict__ BCd, const float* __restrict__ logA,
    const float* __restrict__ Dp, float* __restrict__ y) {
  int t = blockIdx.x * blockDim.x + threadIdx.x;   // DI_*NS_
  int d = t >> 4, n = t & 15;
  float A  = -__expf(logA[d * NS_ + n]);
  float dp = Dp[d];
  float h  = 0.f;
  for (int l = 0; l < L_; ++l) {
    float dl = delta[(size_t)l * DI_ + d];
    float xv = xs[(size_t)l * DI_ + d];
    const float* bc = BCd + (size_t)l * NR_;
    float Bn = bc[DR_ + n];
    float Cn = bc[DR_ + NS_ + n];
    h = __expf(dl * A) * h + (dl * xv) * Bn;
    float p = h * Cn;
    p += __shfl_xor(p, 1);
    p += __shfl_xor(p, 2);
    p += __shfl_xor(p, 4);
    p += __shfl_xor(p, 8);
    if (n == 0) y[(size_t)l * DI_ + d] = p + xv * dp;
  }
}

// y * silu(res) -> f16 (res = second half of xz row)
__global__ void gate_kernel(const float* __restrict__ y,
                            const float* __restrict__ xz,
                            _Float16* __restrict__ yh) {
  int t = blockIdx.x * blockDim.x + threadIdx.x;   // L_*DI_
  int d = t % DI_, l = t / DI_;
  float res = xz[(size_t)l * XZ_ + DI_ + d];
  yh[t] = (_Float16)(y[t] * silu_f(res));
}

__global__ void add_kernel(float* __restrict__ x, const float* __restrict__ t_) {
  int t = blockIdx.x * blockDim.x + threadIdx.x;   // L_*D_
  x[t] += t_[t];
}

// ---------------------------------------------------------------------------
// Host-side orchestration
// ---------------------------------------------------------------------------
static inline void* bump(char* ws, size_t& off, size_t bytes) {
  void* p = ws + off;
  off = (off + bytes + 255) & ~(size_t)255;
  return p;
}

extern "C" void kernel_launch(void* const* d_in, const int* in_sizes, int n_in,
                              void* d_out, int out_size, void* d_ws, size_t ws_size,
                              hipStream_t stream) {
  const int*   tokens      = (const int*)  d_in[0];
  const float* emb         = (const float*)d_in[1];
  const float* rms_w       = (const float*)d_in[2];
  const float* in_proj_w   = (const float*)d_in[3];
  const float* conv_w      = (const float*)d_in[4];
  const float* conv_b      = (const float*)d_in[5];
  const float* xproj_w     = (const float*)d_in[6];
  const float* dt_w        = (const float*)d_in[7];
  const float* dt_b        = (const float*)d_in[8];
  const float* logA        = (const float*)d_in[9];
  const float* Dp          = (const float*)d_in[10];
  const float* out_proj_w  = (const float*)d_in[11];
  const float* final_rms_w = (const float*)d_in[12];
  float* out = (float*)d_out;

  char* ws = (char*)d_ws;
  size_t off = 0;
  float*    x       = (float*)   bump(ws, off, (size_t)L_ * D_  * 4);
  _Float16* xn_h    = (_Float16*)bump(ws, off, (size_t)L_ * D_  * 2);
  float*    xz      = (float*)   bump(ws, off, (size_t)L_ * XZ_ * 4);
  float*    xi_s    = (float*)   bump(ws, off, (size_t)L_ * DI_ * 4);
  _Float16* xi_h    = (_Float16*)bump(ws, off, (size_t)L_ * DI_ * 2);
  float*    BCd     = (float*)   bump(ws, off, (size_t)L_ * NR_ * 4);
  float*    delta   = (float*)   bump(ws, off, (size_t)L_ * DI_ * 4);
  float*    yb      = (float*)   bump(ws, off, (size_t)L_ * DI_ * 4);
  _Float16* yg_h    = (_Float16*)bump(ws, off, (size_t)L_ * DI_ * 2);
  float*    tmp     = (float*)   bump(ws, off, (size_t)L_ * D_  * 4);
  _Float16* w_in_h  = (_Float16*)bump(ws, off, (size_t)XZ_ * D_  * 2);
  _Float16* w_xp_h  = (_Float16*)bump(ws, off, (size_t)NR_ * DI_ * 2);
  _Float16* w_out_h = (_Float16*)bump(ws, off, (size_t)D_  * DI_ * 2);
  _Float16* emb_h   = (_Float16*)bump(ws, off, (size_t)NV_ * D_  * 2);

  const int TPB = 256;
  auto nb = [](int n) { return (n + 255) / 256; };

  // embedding gather + one-time emb -> f16 conversion (reused at the end)
  embed_kernel<<<nb(L_ * D_), TPB, 0, stream>>>(tokens, emb, x);
  cvt_f16_kernel<<<nb(NV_ * D_), TPB, 0, stream>>>(emb, emb_h, NV_ * D_);

  for (int layer = 0; layer < 2; ++layer) {
    const float* inw = in_proj_w  + (size_t)layer * XZ_ * D_;
    const float* cw  = conv_w     + (size_t)layer * DI_ * 4;
    const float* cb  = conv_b     + (size_t)layer * DI_;
    const float* xpw = xproj_w    + (size_t)layer * NR_ * DI_;
    const float* dtw = dt_w       + (size_t)layer * DI_ * DR_;
    const float* dtb = dt_b       + (size_t)layer * DI_;
    const float* lA  = logA       + (size_t)layer * DI_ * NS_;
    const float* dpv = Dp         + (size_t)layer * DI_;
    const float* ow  = out_proj_w + (size_t)layer * D_ * DI_;
    const float* rw  = rms_w      + (size_t)layer * D_;

    // rmsnorm -> f16 activations
    rmsnorm_f16_kernel<<<L_, TPB, 0, stream>>>(x, rw, xn_h);

    // in_proj:  xz[L,3072] = xn @ in_w^T           (WMMA, K=768)
    cvt_f16_kernel<<<nb(XZ_ * D_), TPB, 0, stream>>>(inw, w_in_h, XZ_ * D_);
    gemm_f16_wmma<<<dim3((XZ_ + 127) / 128, L_ / 64), TPB, 0, stream>>>(
        xn_h, w_in_h, xz, L_, XZ_, D_);

    // depthwise causal conv + SiLU
    conv_silu_kernel<<<nb(L_ * DI_), TPB, 0, stream>>>(xz, cw, cb, xi_s, xi_h);

    // x_proj:  BCd[L,80] = xi @ xproj^T            (WMMA, K=1536)
    cvt_f16_kernel<<<nb(NR_ * DI_), TPB, 0, stream>>>(xpw, w_xp_h, NR_ * DI_);
    gemm_f16_wmma<<<dim3((NR_ + 127) / 128, L_ / 64), TPB, 0, stream>>>(
        xi_h, w_xp_h, BCd, L_, NR_, DI_);

    // delta = softplus(dlt @ dt_w^T + dt_b)
    delta_kernel<<<nb(L_ * DI_), TPB, 0, stream>>>(BCd, dtw, dtb, delta);

    // selective scan (lane per (d,n), sequential over L)
    scan_kernel<<<(DI_ * NS_) / TPB, TPB, 0, stream>>>(delta, xi_s, BCd, lA,
                                                       dpv, yb);

    // gate with silu(res)
    gate_kernel<<<nb(L_ * DI_), TPB, 0, stream>>>(yb, xz, yg_h);

    // out_proj:  tmp[L,768] = yg @ out_w^T         (WMMA, K=1536)
    cvt_f16_kernel<<<nb(D_ * DI_), TPB, 0, stream>>>(ow, w_out_h, D_ * DI_);
    gemm_f16_wmma<<<dim3((D_ + 127) / 128, L_ / 64), TPB, 0, stream>>>(
        yg_h, w_out_h, tmp, L_, D_, DI_);

    // residual
    add_kernel<<<nb(L_ * D_), TPB, 0, stream>>>(x, tmp);
  }

  // final rmsnorm + logits GEMM:  out[L,V] = xf @ emb^T   (WMMA, K=768)
  rmsnorm_f16_kernel<<<L_, TPB, 0, stream>>>(x, final_rms_w, xn_h);
  gemm_f16_wmma<<<dim3((NV_ + 127) / 128, L_ / 64), TPB, 0, stream>>>(
      xn_h, emb_h, out, L_, NV_, D_);
}
